// FlashModel_53042846106133
// MI455X (gfx1250) — matrix operand order repaired and checked by reading, not compiled, not run on hardware
//
#include <hip/hip_runtime.h>
#include <hip/hip_bf16.h>

// FLASH/GAU block for MI455X (gfx1250, wave32, WMMA).
// Single bf16 WMMA GEMM (v_wmma_f32_16x16x32_bf16, f32 accumulate),
// software-pipelined LDS staging (K=64/stage, 8 WMMAs per barrier pair),
// padded LDS rows (72 elems) for bank spread. All B operands stored
// N-major by their producers so staging and fragment reads are pure
// b128 LDS traffic (no scalar transpose chains); bf16 GEMM outputs are
// written transposed as 16B vector stores (lane holds 8 consecutive rows).

typedef __attribute__((ext_vector_type(8)))  __bf16 v8bf;
typedef __attribute__((ext_vector_type(16))) __bf16 v16bf;
typedef __attribute__((ext_vector_type(8)))  float  v8f;

#define BB 8
#define GG 16
#define NN 256
#define DD 768
#define EE 1536
#define SS 128
#define TT (BB * GG * NN)      // 32768 tokens

#define LDS_PAD 72             // 144B row stride: 16B-aligned, 36-dword bank spread

// -------------------- generic WMMA GEMM --------------------
// C[M,N] = alpha * A[M,K] @ B'[K,N]  where B is stored N-major [N][K] (ldb = K-length).
// block tile 64(M) x 128(N), 8 waves (2x4), each wave 32x32 via 2x2 frags.
// M%64==0, N%128==0, K%64==0 for all uses.

struct GemmP {
  const __bf16* A; const __bf16* B;
  long long sA, sB, sC;            // batch offsets (elements)
  int M, N, K, lda, ldb, ldc;
  float alpha;
  const float* bias;               // per-col bias
  float* Cf; __bf16* Cb;
  const float* relw;               // rel-pos bias weights (len 1023)
  const float* resid;              // residual (inputs)
  __bf16* u_out; __bf16* v_out; float* z_out;  // GEMM1 split
};

enum { MODE_G1 = 0, MODE_BF16T = 1, MODE_F32_STORE = 2, MODE_F32_ACC = 3,
       MODE_RELU2 = 4, MODE_FINAL = 5 };

__device__ __forceinline__ void ga_load(const __bf16* A, int lda, int bm, int k0,
                                        int tid, int K, v8bf& r0, v8bf& r1) {
  const __bf16* s = A + (long long)(bm + (tid >> 2)) * lda + k0 + (tid & 3) * 16;
  if (k0 + 64 < K) __builtin_prefetch((const void*)(s + 64), 0, 0);
  r0 = *(const v8bf*)s; r1 = *(const v8bf*)(s + 8);
}

__device__ __forceinline__ void ga_store(__bf16 (*lds_a)[LDS_PAD], int tid,
                                         const v8bf& r0, const v8bf& r1) {
  int r = tid >> 2, c = (tid & 3) * 16;
  *(v8bf*)&lds_a[r][c]     = r0;
  *(v8bf*)&lds_a[r][c + 8] = r1;
}

__device__ __forceinline__ void gb_load(const __bf16* B, int ldb, int bn, int k0,
                                        int tid, int K,
                                        v8bf& r0, v8bf& r1, v8bf& r2, v8bf& r3) {
  // B stored N-major: row = n, 32 contiguous K elems per thread
  const __bf16* s = B + (long long)(bn + (tid >> 1)) * ldb + k0 + (tid & 1) * 32;
  if (k0 + 64 < K) __builtin_prefetch((const void*)(s + 64), 0, 0);
  r0 = *(const v8bf*)s;        r1 = *(const v8bf*)(s + 8);
  r2 = *(const v8bf*)(s + 16); r3 = *(const v8bf*)(s + 24);
}

__device__ __forceinline__ void gb_store(__bf16 (*lds_bt)[LDS_PAD], int tid,
                                         const v8bf& r0, const v8bf& r1,
                                         const v8bf& r2, const v8bf& r3) {
  int nn = tid >> 1, kk = (tid & 1) * 32;
  *(v8bf*)&lds_bt[nn][kk]      = r0;
  *(v8bf*)&lds_bt[nn][kk + 8]  = r1;
  *(v8bf*)&lds_bt[nn][kk + 16] = r2;
  *(v8bf*)&lds_bt[nn][kk + 24] = r3;
}

template<int MODE>
__global__ __launch_bounds__(256) void gemm_wmma(GemmP p) {
  __shared__ __align__(16) __bf16 lds_a[64][LDS_PAD];    // [m][k]
  __shared__ __align__(16) __bf16 lds_bt[128][LDS_PAD];  // [n][k]

  const int tid  = threadIdx.x;
  const int lane = tid & 31;
  const int wave = tid >> 5;
  const int wm = wave & 1;          // 0..1
  const int wn = wave >> 1;         // 0..3
  const int bm = blockIdx.x * 64;
  const int bn = blockIdx.y * 128;
  const long long bz = blockIdx.z;

  const __bf16* A = p.A + bz * p.sA;
  const __bf16* B = p.B + bz * p.sB;

  v8f acc[2][2];
#pragma unroll
  for (int i = 0; i < 2; ++i)
#pragma unroll
    for (int j = 0; j < 2; ++j)
#pragma unroll
      for (int e = 0; e < 8; ++e) acc[i][j][e] = 0.f;

  v8bf ra0, ra1, rb0, rb1, rb2, rb3;
  ga_load(A, p.lda, bm, 0, tid, p.K, ra0, ra1);
  gb_load(B, p.ldb, bn, 0, tid, p.K, rb0, rb1, rb2, rb3);

  const int mrow = lane & 15;
  const int kh = (lane >> 4) * 8;    // A: K half select
  const int kb = (lane >> 4) * 16;   // unused (B reads also use kh)

  for (int k0 = 0; k0 < p.K; k0 += 64) {
    ga_store(lds_a, tid, ra0, ra1);
    gb_store(lds_bt, tid, rb0, rb1, rb2, rb3);
    __syncthreads();

    // software pipeline: next tile's global loads overlap the WMMAs below
    if (k0 + 64 < p.K) {
      ga_load(A, p.lda, bm, k0 + 64, tid, p.K, ra0, ra1);
      gb_load(B, p.ldb, bn, k0 + 64, tid, p.K, rb0, rb1, rb2, rb3);
    }

#pragma unroll
    for (int ks = 0; ks < 64; ks += 32) {
      v16bf af[2], bfr[2];
#pragma unroll
      for (int f = 0; f < 2; ++f) {
        int m = wm * 32 + f * 16 + mrow;
        v8bf lo = *(const v8bf*)&lds_a[m][ks + kh];
        v8bf hi = *(const v8bf*)&lds_a[m][ks + 16 + kh];
#pragma unroll
        for (int j = 0; j < 8; ++j) { af[f][j] = lo[j]; af[f][8 + j] = hi[j]; }
      }
#pragma unroll
      for (int f = 0; f < 2; ++f) {
        int n = wn * 32 + f * 16 + mrow;
        v8bf lo = *(const v8bf*)&lds_bt[n][ks + kh];
        v8bf hi = *(const v8bf*)&lds_bt[n][ks + 16 + kh];
#pragma unroll
        for (int j = 0; j < 8; ++j) { bfr[f][j] = lo[j]; bfr[f][8 + j] = hi[j]; }
      }
#pragma unroll
      for (int i = 0; i < 2; ++i)
#pragma unroll
        for (int j = 0; j < 2; ++j)
          acc[i][j] = __builtin_amdgcn_wmma_f32_16x16x32_bf16(
              false, af[i], false, bfr[j], (short)0, acc[i][j], false, false);
    }
    __syncthreads();
  }

  // ---- epilogue (per 16x16 fragment: lane owns column c, rows r0..r0+7)
  const int ccol0 = bn + wn * 32 + (lane & 15);
  const int rsub  = (lane >> 4) * 8;
#pragma unroll
  for (int i = 0; i < 2; ++i) {
#pragma unroll
    for (int j = 0; j < 2; ++j) {
      int c  = ccol0 + j * 16;
      int r0 = bm + wm * 32 + i * 16 + rsub;
      if (MODE == MODE_BF16T) {
        // transposed N-major store: one 16B vector per fragment
        v8bf o;
#pragma unroll
        for (int e = 0; e < 8; ++e) o[e] = (__bf16)(acc[i][j][e] * p.alpha);
        *(v8bf*)&p.Cb[(long long)c * p.ldc + bz * p.sC + r0] = o;
      } else if (MODE == MODE_G1) {
        float bcol = p.bias[c];
        if (c < EE) {
#pragma unroll
          for (int e = 0; e < 8; ++e) {
            float v = acc[i][j][e] + bcol;
            float s = v * __builtin_amdgcn_rcpf(1.f + __expf(-v));
            p.u_out[(long long)(r0 + e) * EE + c] = (__bf16)s;   // row-major (A of GEMM2)
          }
        } else if (c < 2 * EE) {
          v8bf o;
#pragma unroll
          for (int e = 0; e < 8; ++e) {
            float v = acc[i][j][e] + bcol;
            o[e] = (__bf16)(v * __builtin_amdgcn_rcpf(1.f + __expf(-v)));
          }
          *(v8bf*)&p.v_out[(long long)(c - EE) * TT + r0] = o;   // N-major (B operand)
        } else {
#pragma unroll
          for (int e = 0; e < 8; ++e) {
            float v = acc[i][j][e] + bcol;
            p.z_out[(long long)(r0 + e) * SS + (c - 2 * EE)] =
                v * __builtin_amdgcn_rcpf(1.f + __expf(-v));
          }
        }
      } else {
#pragma unroll
        for (int e = 0; e < 8; ++e) {
          int r = r0 + e;
          float v = acc[i][j][e] * p.alpha;
          if (MODE == MODE_F32_STORE) {
            p.Cf[bz * p.sC + (long long)r * p.ldc + c] = v;
          } else if (MODE == MODE_F32_ACC) {
            p.Cf[bz * p.sC + (long long)r * p.ldc + c] += v;
          } else if (MODE == MODE_RELU2) {
            float x = v + p.relw[511 + c - r];      // bias[n,m]=w_rel[511+m-n]
            x = x > 0.f ? x : 0.f;
            p.Cb[bz * p.sC + (long long)r * p.ldc + c] = (__bf16)(x * x);
          } else { // MODE_FINAL
            long long idx = (long long)r * p.ldc + c;
            p.Cf[idx] = v + p.bias[c] + p.resid[idx];
          }
        }
      }
    }
  }
}

// -------------------- elementwise / small kernels --------------------

__global__ __launch_bounds__(256) void cvt_t_kernel(const float* __restrict__ a,
                                                    __bf16* __restrict__ o,
                                                    int K, int N) {
  // a: [K][N] row-major  ->  o: [N][K] (N-major for B staging)
  long long i = (long long)blockIdx.x * 256 + threadIdx.x;
  if (i >= (long long)K * N) return;
  int n = (int)(i / K), k = (int)(i % K);
  o[i] = (__bf16)a[(long long)k * N + n];
}

__global__ __launch_bounds__(256) void ln_kernel(const float* __restrict__ x,
                                                 const float* __restrict__ w,
                                                 const float* __restrict__ b,
                                                 __bf16* __restrict__ out) {
  __shared__ float r1[8], r2[8];
  long long row = blockIdx.x;
  const float* xr = x + row * DD;
  float s1 = 0.f, s2 = 0.f;
  for (int c = threadIdx.x; c < DD; c += 256) { float v = xr[c]; s1 += v; s2 += v * v; }
#pragma unroll
  for (int o = 16; o > 0; o >>= 1) { s1 += __shfl_down(s1, o, 32); s2 += __shfl_down(s2, o, 32); }
  int lane = threadIdx.x & 31, wv = threadIdx.x >> 5;
  if (lane == 0) { r1[wv] = s1; r2[wv] = s2; }
  __syncthreads();
  if (threadIdx.x == 0) {
    float a = 0.f, c2 = 0.f;
    for (int i = 0; i < 8; ++i) { a += r1[i]; c2 += r2[i]; }
    r1[0] = a; r2[0] = c2;
  }
  __syncthreads();
  float mu  = r1[0] * (1.f / DD);
  float var = r2[0] * (1.f / DD) - mu * mu;
  float inv = rsqrtf(var + 1e-5f);
  __bf16* orow = out + row * DD;
  for (int c = threadIdx.x; c < DD; c += 256)
    orow[c] = (__bf16)((xr[c] - mu) * inv * w[c] + b[c]);
}

__global__ __launch_bounds__(256) void rope_kernel(const float* __restrict__ zf,
                                                   const float* __restrict__ gamma,
                                                   const float* __restrict__ beta,
                                                   __bf16* __restrict__ q0,   // quad_q  [T][S]
                                                   __bf16* __restrict__ q1,   // quad_k  [T][S] (B N-major)
                                                   __bf16* __restrict__ q2,   // lin_q   [T][S]
                                                   __bf16* __restrict__ q3t) {// lin_k^T [S][T]
  long long t = blockIdx.x;            // token 0..32767
  int head = threadIdx.x >> 6;         // 0..3
  int k = threadIdx.x & 63;            // 0..63 (half)
  int pos = (int)(t & (GG * NN - 1));  // position = t % (G*N)
  const float* z = zf + t * SS;
  float x1 = z[k]      * gamma[head * SS + k]      + beta[head * SS + k];
  float x2 = z[64 + k] * gamma[head * SS + 64 + k] + beta[head * SS + 64 + k];
  float fr = __powf(10000.f, (float)k * (1.f / 64.f));
  float ang = (float)pos * fr;
  float sn = __sinf(ang), cs = __cosf(ang);
  float o1 = x1 * cs - x2 * sn;
  float o2 = x2 * cs + x1 * sn;
  if (head == 3) {           // lin_k stored transposed [S][T]
    q3t[(long long)k * TT + t]        = (__bf16)o1;
    q3t[(long long)(64 + k) * TT + t] = (__bf16)o2;
  } else {
    __bf16* outs[3] = { q0, q1, q2 };
    __bf16* o = outs[head] + t * SS;
    o[k]      = (__bf16)o1;
    o[64 + k] = (__bf16)o2;
  }
}

__global__ __launch_bounds__(256) void mask_kernel(const int* __restrict__ seg,
                                                   float* __restrict__ mask) {
  int b = blockIdx.x;
  __shared__ int mn[GG], mx[GG];
  __shared__ float ov[GG][GG];
  int g = threadIdx.x >> 4;
  int sub = threadIdx.x & 15;
  if (sub == 0) { mn[g] = 0x7fffffff; mx[g] = (int)0x80000000; }
  __syncthreads();
  const int* s = seg + (long long)b * (GG * NN) + (long long)g * NN;
  int lmn = 0x7fffffff, lmx = (int)0x80000000;
  for (int j = sub; j < NN; j += 16) { int v = s[j]; lmn = min(lmn, v); lmx = max(lmx, v); }
  atomicMin(&mn[g], lmn);
  atomicMax(&mx[g], lmx);
  __syncthreads();
  int h = sub;
  float o = (mn[g] <= mx[h] && mx[g] >= mn[h]) ? 1.f : 0.f;
  ov[g][h] = o;
  __syncthreads();
  float sum = 0.f;
#pragma unroll
  for (int j = 0; j < GG; ++j) sum += ov[g][j];
  mask[(long long)b * GG * GG + g * GG + h] = o / sum;
}

__global__ __launch_bounds__(256) void mix_kernel(const float* __restrict__ linkv,
                                                  const float* __restrict__ mask,
                                                  __bf16* __restrict__ outT) {
  // linkv: [b][g][s][e] f32 (row-major), outT: [e][(b*G+g)*S + s] bf16 (N-major)
  const long long per_b = (long long)GG * SS * EE;   // 3,145,728
  const long long per_g = (long long)SS * EE;        // 196,608
  long long i = (long long)blockIdx.x * 256 + threadIdx.x;
  if (i >= (long long)BB * per_b) return;
  int b = (int)(i / per_b);
  long long rb = i - (long long)b * per_b;
  int g = (int)(rb / per_g);
  long long ke = rb - (long long)g * per_g;
  const float* m = mask + (long long)b * GG * GG + g * GG;
  const float* src = linkv + (long long)b * per_b + ke;
  float s = 0.f;
#pragma unroll
  for (int h = 0; h < GG; ++h) s += m[h] * src[(long long)h * per_g];
  int e = (int)(ke % EE), srow = (int)(ke / EE);
  outT[(long long)e * (BB * GG * SS) + (long long)(b * GG + g) * SS + srow] = (__bf16)s;
}

__global__ __launch_bounds__(256) void umul_kernel(__bf16* __restrict__ u,
                                                   const float* __restrict__ acc, long long n) {
  long long i = (long long)blockIdx.x * 256 + threadIdx.x;
  if (i < n) u[i] = (__bf16)((float)u[i] * acc[i]);
}

// -------------------- host orchestration --------------------

extern "C" void kernel_launch(void* const* d_in, const int* in_sizes, int n_in,
                              void* d_out, int out_size, void* d_ws, size_t ws_size,
                              hipStream_t stream) {
  (void)in_sizes; (void)n_in; (void)out_size; (void)ws_size;
  const float* inputs = (const float*)d_in[0];
  const int*   seg    = (const int*)d_in[1];
  const float* ln_w   = (const float*)d_in[2];
  const float* ln_b   = (const float*)d_in[3];
  const float* W1     = (const float*)d_in[4];
  const float* b1     = (const float*)d_in[5];
  const float* gamma  = (const float*)d_in[6];
  const float* beta   = (const float*)d_in[7];
  const float* W2     = (const float*)d_in[8];
  const float* b2     = (const float*)d_in[9];
  const float* w_rel  = (const float*)d_in[10];
  float* out = (float*)d_out;

  size_t off = 0;
  auto alloc = [&](size_t bytes) -> void* {
    void* p = (char*)d_ws + off;
    off = (off + bytes + 255) & ~(size_t)255;
    return p;
  };
  __bf16* xb     = (__bf16*)alloc((size_t)TT * DD * 2);
  __bf16* w1t    = (__bf16*)alloc((size_t)DD * (2 * EE + SS) * 2);   // [3200][768]
  __bf16* w2t    = (__bf16*)alloc((size_t)EE * DD * 2);              // [768][1536]
  __bf16* ub     = (__bf16*)alloc((size_t)TT * EE * 2);              // [T][E] row-major
  __bf16* vbT    = (__bf16*)alloc((size_t)TT * EE * 2);              // [E][T] N-major
  float*  zf     = (float*) alloc((size_t)TT * SS * 4);
  __bf16* qq     = (__bf16*)alloc((size_t)TT * SS * 2);              // quad_q [T][S]
  __bf16* qk     = (__bf16*)alloc((size_t)TT * SS * 2);              // quad_k [T][S]
  __bf16* lq     = (__bf16*)alloc((size_t)TT * SS * 2);              // lin_q  [T][S]
  __bf16* lkT    = (__bf16*)alloc((size_t)TT * SS * 2);              // lin_k^T [S][T]
  __bf16* kvT    = (__bf16*)alloc((size_t)BB * SS * EE * 2);         // [E][B*S]
  __bf16* linvT  = (__bf16*)alloc((size_t)TT * EE * 2);              // [E][T]
  float*  linkv  = (float*) alloc((size_t)BB * GG * SS * EE * 4);    // [b][g][s][e]
  __bf16* linkvT = (__bf16*)alloc((size_t)BB * GG * SS * EE * 2);    // [E][B*G*S]
  float*  maskb  = (float*) alloc((size_t)BB * GG * GG * 4);
  float*  accb   = (float*) alloc((size_t)TT * EE * 4);
  __bf16* kern   = xb;   // reuse: xb dead after GEMM1 (16.8MB <= 50MB)

  // 0) transpose-convert weights to bf16 N-major
  {
    long long n1 = (long long)DD * (2 * EE + SS);
    cvt_t_kernel<<<dim3((unsigned)((n1 + 255) / 256)), 256, 0, stream>>>(W1, w1t, DD, 2 * EE + SS);
    long long n2 = (long long)EE * DD;
    cvt_t_kernel<<<dim3((unsigned)((n2 + 255) / 256)), 256, 0, stream>>>(W2, w2t, EE, DD);
  }
  // 1) LayerNorm -> xb (bf16)
  ln_kernel<<<dim3((unsigned)TT), 256, 0, stream>>>(inputs, ln_w, ln_b, xb);

  // 2) GEMM1: silu(xb @ W1 + b1) -> u row-major, v N-major, z f32
  {
    GemmP p{}; p.A = xb; p.B = w1t; p.M = TT; p.N = 2 * EE + SS; p.K = DD;
    p.lda = DD; p.ldb = DD; p.alpha = 1.f; p.bias = b1;
    p.u_out = ub; p.v_out = vbT; p.z_out = zf;
    gemm_wmma<MODE_G1><<<dim3(512, 25, 1), 256, 0, stream>>>(p);
  }
  // 3) gamma/beta + RoPE
  rope_kernel<<<dim3((unsigned)TT), 256, 0, stream>>>(zf, gamma, beta, qq, qk, lq, lkT);

  // 4) kv[b] = lin_k[b]^T @ v[b]   (S x E, K = G*N = 4096) -> kvT [E][B*S]
  {
    GemmP p{}; p.A = lkT; p.B = vbT; p.Cb = kvT;
    p.M = SS; p.N = EE; p.K = GG * NN;
    p.lda = TT; p.ldb = TT; p.ldc = BB * SS; p.alpha = 1.f;
    p.sA = GG * NN; p.sB = GG * NN; p.sC = SS;    // column offsets per batch
    gemm_wmma<MODE_BF16T><<<dim3(2, 12, BB), 256, 0, stream>>>(p);
  }
  // 5) lin_v[b] = lin_q[b] @ kv[b]  (fold g into M: 4096 x E, K = S) -> linvT [E][T]
  {
    GemmP p{}; p.A = lq; p.B = kvT; p.Cb = linvT;
    p.M = GG * NN; p.N = EE; p.K = SS;
    p.lda = SS; p.ldb = BB * SS; p.ldc = TT; p.alpha = 1.f;
    p.sA = (long long)GG * NN * SS; p.sB = SS; p.sC = GG * NN;
    gemm_wmma<MODE_BF16T><<<dim3(64, 12, BB), 256, 0, stream>>>(p);
  }
  // 6) lin_kv[b,g] = lin_k[b,g]^T @ lin_v[b,g] / N  (S x E, K = N) -> linkv f32 row-major
  {
    GemmP p{}; p.A = lkT; p.B = linvT; p.Cf = linkv;
    p.M = SS; p.N = EE; p.K = NN;
    p.lda = TT; p.ldb = TT; p.ldc = EE; p.alpha = 1.f / (float)NN;
    p.sA = NN; p.sB = NN; p.sC = (long long)SS * EE;
    gemm_wmma<MODE_F32_STORE><<<dim3(2, 12, BB * GG), 256, 0, stream>>>(p);
  }
  // 7) segment-overlap mask + group mixing -> linkvT [E][B*G*S]
  mask_kernel<<<dim3(BB), 256, 0, stream>>>(seg, maskb);
  {
    long long n = (long long)BB * GG * SS * EE;
    mix_kernel<<<dim3((unsigned)((n + 255) / 256)), 256, 0, stream>>>(linkv, maskb, linkvT);
  }
  // 8) linear[b,g] = lin_q[b,g] @ lin_kv_mixed[b,g] -> acc (f32 store, row-major)
  {
    GemmP p{}; p.A = lq; p.B = linkvT; p.Cf = accb;
    p.M = NN; p.N = EE; p.K = SS;
    p.lda = SS; p.ldb = BB * GG * SS; p.ldc = EE; p.alpha = 1.f;
    p.sA = (long long)NN * SS; p.sB = SS; p.sC = (long long)NN * EE;
    gemm_wmma<MODE_F32_STORE><<<dim3(4, 12, BB * GG), 256, 0, stream>>>(p);
  }
  // 9) kernel[b,g] = relu(quad_q @ quad_k^T / N + relbias)^2 -> bf16 row-major
  {
    GemmP p{}; p.A = qq; p.B = qk; p.Cb = kern; p.relw = w_rel;
    p.M = NN; p.N = NN; p.K = SS;
    p.lda = SS; p.ldb = SS; p.ldc = NN; p.alpha = 1.f / (float)NN;
    p.sA = (long long)NN * SS; p.sB = (long long)NN * SS; p.sC = (long long)NN * NN;
    gemm_wmma<MODE_RELU2><<<dim3(4, 2, BB * GG), 256, 0, stream>>>(p);
  }
  // 10) acc += kernel @ v   (quadratic branch; vbT columns bg*256..)
  {
    GemmP p{}; p.A = kern; p.B = vbT; p.Cf = accb;
    p.M = NN; p.N = EE; p.K = NN;
    p.lda = NN; p.ldb = TT; p.ldc = EE; p.alpha = 1.f;
    p.sA = (long long)NN * NN; p.sB = NN; p.sC = (long long)NN * EE;
    gemm_wmma<MODE_F32_ACC><<<dim3(4, 12, BB * GG), 256, 0, stream>>>(p);
  }
  // 11) u *= (quadratic + linear)  (in place, bf16)
  {
    long long n = (long long)TT * EE;
    umul_kernel<<<dim3((unsigned)((n + 255) / 256)), 256, 0, stream>>>(ub, accb, n);
  }
  // 12) out = (u*(quad+lin)) @ W2 + b2 + shortcut
  {
    GemmP p{}; p.A = ub; p.B = w2t; p.Cf = out; p.bias = b2; p.resid = inputs;
    p.M = TT; p.N = DD; p.K = EE;
    p.lda = EE; p.ldb = EE; p.ldc = DD; p.alpha = 1.f;
    gemm_wmma<MODE_FINAL><<<dim3(512, 6, 1), 256, 0, stream>>>(p);
  }
}
